// Head_21414706937885
// MI455X (gfx1250) — compile-verified, matrix operand order checked by reading, not run on hardware
//
#include <hip/hip_runtime.h>
#include <hip/hip_bf16.h>

// ---------------------------------------------------------------------------
// Causal attention head, flash-attention style, f16 WMMA (gfx1250 / CDNA5).
// B=4, T=4096, C=1024, H=64.
// ---------------------------------------------------------------------------

#define Bn 4
#define Tn 4096
#define Cn 1024
#define Hn 64

typedef _Float16 h8   __attribute__((ext_vector_type(8)));
typedef _Float16 v16h __attribute__((ext_vector_type(16)));
typedef float    v8f  __attribute__((ext_vector_type(8)));
typedef float    f4   __attribute__((ext_vector_type(4)));

static __device__ inline v8f wmma16(v16h a, v16h b, v8f c) {
  // D = A(16x32 f16) * B(32x16 f16) + C(16x16 f32)
  return __builtin_amdgcn_wmma_f32_16x16x32_f16(
      /*neg_a=*/false, a, /*neg_b=*/false, b,
      /*c_mod=*/(short)0, c, /*reuse_a=*/false, /*reuse_b=*/false);
}

// Load a 16x32 f16 A-fragment (also serves as the B-fragment of src^T):
// lane&15 selects the M row; lane>>4 selects K-chunks {0..7,16..23} vs
// {8..15,24..31}; halves [0..7] and [8..15] are two contiguous b128 loads.
static __device__ inline v16h load_frag16(const _Float16* __restrict__ src,
                                          int row_base, int ld, int col0,
                                          int lane) {
  int n = lane & 15, hh = lane >> 4;
  const _Float16* p = src + (size_t)(row_base + n) * ld + col0 + hh * 8;
  h8 lo = *(const h8*)p;
  h8 hi = *(const h8*)(p + 16);
  return __builtin_shufflevector(lo, hi, 0, 1, 2, 3, 4, 5, 6, 7, 8, 9, 10, 11,
                                 12, 13, 14, 15);
}

// Same fragment shape but from an f32 source (x), converting to f16.
static __device__ inline v16h load_frag_f32(const float* __restrict__ src,
                                            int row_base, int ld, int col0,
                                            int lane) {
  int n = lane & 15, hh = lane >> 4;
  const float* p = src + (size_t)(row_base + n) * ld + col0 + hh * 8;
  f4 a0 = *(const f4*)p;
  f4 a1 = *(const f4*)(p + 4);
  f4 b0 = *(const f4*)(p + 16);
  f4 b1 = *(const f4*)(p + 20);
  v16h r;
#pragma unroll
  for (int i = 0; i < 4; i++) {
    r[i]      = (_Float16)a0[i];
    r[4 + i]  = (_Float16)a1[i];
    r[8 + i]  = (_Float16)b0[i];
    r[12 + i] = (_Float16)b1[i];
  }
  return r;
}

// ---------------------------------------------------------------------------
// Kernel 1: wT[m][h][c] = w_m[c][h] as f16, m: 0=q,1=k,2=v.
// ---------------------------------------------------------------------------
__global__ void wt_transpose_kernel(const float* __restrict__ wk,
                                    const float* __restrict__ wq,
                                    const float* __restrict__ wv,
                                    _Float16* __restrict__ wT) {
  int idx = blockIdx.x * blockDim.x + threadIdx.x;  // 3*H*C threads exactly
  int m = idx / (Hn * Cn);
  int r = idx % (Hn * Cn);
  int h = r / Cn;
  int c = r % Cn;
  const float* w = (m == 0) ? wq : (m == 1) ? wk : wv;
  wT[idx] = (_Float16)w[c * Hn + h];
}

// ---------------------------------------------------------------------------
// Kernel 2: projections. One wave = one 16-row tile of x; single pass over
// C with all 12 accumulators (q,k,v x 4 h-tiles) live, so each x fragment is
// loaded + f32->f16 converted exactly once, software-pipelined one K-step
// ahead so loads/converts overlap the WMMAs.
// Produces q,k row-major f16 [B][T][H] and v transposed f16 [B][H][T].
// ---------------------------------------------------------------------------
__global__ void proj_kernel(const float* __restrict__ x,
                            const _Float16* __restrict__ wT,
                            _Float16* __restrict__ q,
                            _Float16* __restrict__ k,
                            _Float16* __restrict__ vT) {
  int lane = threadIdx.x & 31;
  int wave = blockIdx.x * (blockDim.x >> 5) + (threadIdx.x >> 5);
  int b = wave / (Tn / 16);
  int trow = (wave % (Tn / 16)) * 16;
  int n = lane & 15, hh = lane >> 4;
  const float* xb = x + (size_t)b * Tn * Cn;

  v8f acc[12] = {};
  v16h A = load_frag_f32(xb, trow, Cn, 0, lane);
  for (int s = 0; s < Cn / 32; s++) {
    v16h Acur = A;
    if (s + 1 < Cn / 32)  // uniform branch; prefetch next A fragment
      A = load_frag_f32(xb, trow, Cn, (s + 1) * 32, lane);
#pragma unroll
    for (int m = 0; m < 3; m++) {
      const _Float16* w = wT + (size_t)m * Hn * Cn;
#pragma unroll
      for (int t = 0; t < 4; t++) {
        v16h Bf = load_frag16(w, t * 16, Cn, s * 32, lane);
        acc[m * 4 + t] = wmma16(Acur, Bf, acc[m * 4 + t]);
      }
    }
  }

  // C layout: lane holds column h=(16t+n), rows trow + 8*hh + i.
#pragma unroll
  for (int m = 0; m < 2; m++) {
    _Float16* dst = (m == 0 ? q : k) + (size_t)b * Tn * Hn;
#pragma unroll
    for (int t = 0; t < 4; t++)
#pragma unroll
      for (int i = 0; i < 8; i++)
        dst[(size_t)(trow + 8 * hh + i) * Hn + t * 16 + n] =
            (_Float16)acc[m * 4 + t][i];
  }
  {
    _Float16* dst = vT + (size_t)b * Hn * Tn;
#pragma unroll
    for (int t = 0; t < 4; t++) {
      h8 r;
#pragma unroll
      for (int i = 0; i < 8; i++) r[i] = (_Float16)acc[8 + t][i];
      *(h8*)(dst + (size_t)(t * 16 + n) * Tn + trow + 8 * hh) = r;
    }
  }
}

// ---------------------------------------------------------------------------
// Kernel 3: flash attention. One wave = one (b, 16-query tile).
// Computes S^T = K*Q^T per 32-key block; the C-layout of the two S^T tiles
// is, per lane, exactly the B-fragment element set of P^T -> after online
// softmax, only per-element f32->f16 conversion is needed before the
// O^T = V^T * P^T WMMAs (no shuffles / LDS).
// Key loop is software-pipelined: current-block V loads issue first, next-
// block K+bias loads issue right after the S^T WMMAs, and the softmax
// VALU/trans work co-executes while both are in flight.
// ---------------------------------------------------------------------------
__global__ void attn_kernel(const _Float16* __restrict__ q,
                            const _Float16* __restrict__ k,
                            const _Float16* __restrict__ vT,
                            const float* __restrict__ bias,
                            float* __restrict__ out) {
  const float NEG_INF = -__builtin_inff();
  int lane = threadIdx.x & 31;
  int wave = blockIdx.x * (blockDim.x >> 5) + (threadIdx.x >> 5);
  int b = wave / (Tn / 16);
  int qbase = (wave % (Tn / 16)) * 16;
  int n = lane & 15, hh = lane >> 4;

  const _Float16* qb = q + (size_t)b * Tn * Hn;
  const _Float16* kb = k + (size_t)b * Tn * Hn;
  const _Float16* vb = vT + (size_t)b * Hn * Tn;

  // Q^T B-fragments (contraction over h split 0..31 / 32..63), loaded once.
  v16h Bq0 = load_frag16(qb, qbase, Hn, 0, lane);
  v16h Bq1 = load_frag16(qb, qbase, Hn, 32, lane);

  v8f o0 = {}, o1 = {}, o2 = {}, o3 = {};
  float mrun = NEG_INF, lrun = 0.f;
  int qg = qbase + n;                       // this lane's query column
  const float* brow = bias + (size_t)qg * Tn;

  int nkb = (qbase + 47) >> 5;              // causal 32-key block count

  // Preload K fragments + bias for block 0.
  v16h ka0 = load_frag16(kb, 0, Hn, 0, lane);
  v16h ka1 = load_frag16(kb, 0, Hn, 32, lane);
  v16h kc0 = load_frag16(kb, 16, Hn, 0, lane);
  v16h kc1 = load_frag16(kb, 16, Hn, 32, lane);
  const float* bp0 = brow + 8 * hh;
  f4 bA0 = *(const f4*)bp0;
  f4 bA1 = *(const f4*)(bp0 + 4);
  f4 bB0 = *(const f4*)(bp0 + 16);
  f4 bB1 = *(const f4*)(bp0 + 20);

  for (int kbi = 0; kbi < nkb; kbi++) {
    int kbase = kbi * 32;

    // Issue current block's V^T A-fragments early (consumed after softmax).
    v16h vf0 = load_frag16(vb, 0,  Tn, kbase, lane);
    v16h vf1 = load_frag16(vb, 16, Tn, kbase, lane);
    v16h vf2 = load_frag16(vb, 32, Tn, kbase, lane);
    v16h vf3 = load_frag16(vb, 48, Tn, kbase, lane);

    // S^T tiles: rows = keys (tile A: kbase..+15, tile B: +16..+31).
    v8f sA = {}, sB = {};
    sA = wmma16(ka0, Bq0, sA);
    sA = wmma16(ka1, Bq1, sA);
    sB = wmma16(kc0, Bq0, sB);
    sB = wmma16(kc1, Bq1, sB);

    // Prefetch next block's K fragments + bias (uniform branch).
    v16h na0 = ka0, na1 = ka1, nc0 = kc0, nc1 = kc1;
    f4 nA0 = bA0, nA1 = bA1, nB0 = bB0, nB1 = bB1;
    if (kbi + 1 < nkb) {
      int nk = kbase + 32;
      na0 = load_frag16(kb, nk, Hn, 0, lane);
      na1 = load_frag16(kb, nk, Hn, 32, lane);
      nc0 = load_frag16(kb, nk + 16, Hn, 0, lane);
      nc1 = load_frag16(kb, nk + 16, Hn, 32, lane);
      const float* np = brow + nk + 8 * hh;
      nA0 = *(const f4*)np;
      nA1 = *(const f4*)(np + 4);
      nB0 = *(const f4*)(np + 16);
      nB1 = *(const f4*)(np + 20);
    }

    // Scale + bias + causal mask + online softmax (co-executes with loads).
    float svA[8], svB[8];
    float mx = NEG_INF;
#pragma unroll
    for (int i = 0; i < 8; i++) {
      int keyA = kbase + 8 * hh + i;
      float va = sA[i] * 0.125f + ((i < 4) ? bA0[i] : bA1[i - 4]);
      va = (keyA > qg) ? NEG_INF : va;
      svA[i] = va;
      mx = fmaxf(mx, va);
      float vbv = sB[i] * 0.125f + ((i < 4) ? bB0[i] : bB1[i - 4]);
      vbv = (keyA + 16 > qg) ? NEG_INF : vbv;
      svB[i] = vbv;
      mx = fmaxf(mx, vbv);
    }
    // Lanes l and l^16 hold the two key-halves of the same query column.
    mx = fmaxf(mx, __shfl_xor(mx, 16, 32));
    float mnew = fmaxf(mrun, mx);
    float alpha = __expf(mrun - mnew);

    float psum = 0.f;
    v16h P;                                  // B-fragment of P^T, directly
#pragma unroll
    for (int i = 0; i < 8; i++) {
      float pa = __expf(svA[i] - mnew);
      float pb = __expf(svB[i] - mnew);
      psum += pa + pb;
      P[i] = (_Float16)pa;
      P[8 + i] = (_Float16)pb;
    }
    psum += __shfl_xor(psum, 16, 32);
    lrun = lrun * alpha + psum;
    mrun = mnew;

    o0 *= alpha; o1 *= alpha; o2 *= alpha; o3 *= alpha;

    // O^T += V^T * P^T, 4 head-dim tiles, K=32.
    o0 = wmma16(vf0, P, o0);
    o1 = wmma16(vf1, P, o1);
    o2 = wmma16(vf2, P, o2);
    o3 = wmma16(vf3, P, o3);

    // Rotate double buffers.
    ka0 = na0; ka1 = na1; kc0 = nc0; kc1 = nc1;
    bA0 = nA0; bA1 = nA1; bB0 = nB0; bB1 = nB1;
  }

  // Epilogue: out[b][q][h] = O^T / lrun ; per tile t this lane holds 8
  // contiguous h values at h = 16t + 8*hh + i.
  float inv = 1.f / lrun;
  float* op = out + ((size_t)b * Tn + qg) * Hn + 8 * hh;
  v8f oo[4] = {o0, o1, o2, o3};
#pragma unroll
  for (int t = 0; t < 4; t++) {
    f4 lo = {oo[t][0] * inv, oo[t][1] * inv, oo[t][2] * inv, oo[t][3] * inv};
    f4 hi = {oo[t][4] * inv, oo[t][5] * inv, oo[t][6] * inv, oo[t][7] * inv};
    *(f4*)(op + t * 16) = lo;
    *(f4*)(op + t * 16 + 4) = hi;
  }
}

// ---------------------------------------------------------------------------
// Host launcher. Inputs (setup_inputs order): x, head_bias, w_k, w_q, w_v.
// Workspace: q(2MB) | k(2MB) | vT(2MB) | wT(3*128KB).
// ---------------------------------------------------------------------------
extern "C" void kernel_launch(void* const* d_in, const int* in_sizes, int n_in,
                              void* d_out, int out_size, void* d_ws,
                              size_t ws_size, hipStream_t stream) {
  (void)in_sizes; (void)n_in; (void)out_size; (void)ws_size;
  const float* x    = (const float*)d_in[0];
  const float* bias = (const float*)d_in[1];
  const float* wk   = (const float*)d_in[2];
  const float* wq   = (const float*)d_in[3];
  const float* wv   = (const float*)d_in[4];
  float* out = (float*)d_out;

  char* ws = (char*)d_ws;
  _Float16* qbuf = (_Float16*)(ws);
  _Float16* kbuf = (_Float16*)(ws + (size_t)2 * 1024 * 1024);
  _Float16* vTb  = (_Float16*)(ws + (size_t)4 * 1024 * 1024);
  _Float16* wT   = (_Float16*)(ws + (size_t)6 * 1024 * 1024);

  // 3*H*C = 196608 elements
  wt_transpose_kernel<<<768, 256, 0, stream>>>(wk, wq, wv, wT);
  // B*T/16 = 1024 tiles, 8 waves/block
  proj_kernel<<<128, 256, 0, stream>>>(x, wT, qbuf, kbuf, vTb);
  attn_kernel<<<128, 256, 0, stream>>>(qbuf, kbuf, vTb, bias, out);
}
